// MHAttentionWithBias_46608985096472
// MI455X (gfx1250) — matrix operand order repaired and checked by reading, not compiled, not run on hardware
//
#include <hip/hip_runtime.h>

typedef __attribute__((ext_vector_type(16))) _Float16 half16;
typedef __attribute__((ext_vector_type(8)))  _Float16 half8;
typedef __attribute__((ext_vector_type(8)))  float    v8f;

#define DD    512
#define LSEQ  1024
#define NH    8
#define DHEAD 64

// ---------------------------------------------------------------------------
// Kernel 0: one-shot packing.
//  region 0      : X  [4096x512] f32 -> Xh f16 (row major, 8 elems/thread)
//  regions 1..4  : W{q,k,v,o} [512x512] f32 -> f16 TRANSPOSED (Wt[n][k]),
//                  so WMMA B-fragments become contiguous b128 loads.
// Total extra traffic ~12 MB once: negligible vs the 256 MB bias stream.
// ---------------------------------------------------------------------------
__global__ __launch_bounds__(256) void pack_kernel(
    const float* __restrict__ X,
    const float* __restrict__ Wq, const float* __restrict__ Wk,
    const float* __restrict__ Wv, const float* __restrict__ Wo,
    _Float16* __restrict__ Xh,
    _Float16* __restrict__ Wqt, _Float16* __restrict__ Wkt,
    _Float16* __restrict__ Wvt, _Float16* __restrict__ Wot)
{
    const int region = blockIdx.x >> 10;                       // 0..4
    const int tid    = ((blockIdx.x & 1023) << 8) + threadIdx.x;

    if (region == 0) {
        // 2M elements, 8 per thread, fully coalesced both sides
        const float4* src = (const float4*)X + (size_t)tid * 2;
        float4 a = src[0], b = src[1];
        half8 h;
        h[0] = (_Float16)a.x; h[1] = (_Float16)a.y;
        h[2] = (_Float16)a.z; h[3] = (_Float16)a.w;
        h[4] = (_Float16)b.x; h[5] = (_Float16)b.y;
        h[6] = (_Float16)b.z; h[7] = (_Float16)b.w;
        ((half8*)Xh)[tid] = h;
    } else {
        const float* W  = (region == 1) ? Wq : (region == 2) ? Wk
                        : (region == 3) ? Wv : Wo;
        _Float16*   Wt  = (region == 1) ? Wqt : (region == 2) ? Wkt
                        : (region == 3) ? Wvt : Wot;
        const int n = tid >> 9;        // 0..511
        const int k = tid & 511;       // consecutive tid -> coalesced writes
        Wt[(size_t)n * DD + k] = (_Float16)W[(size_t)k * DD + n];
    }
}

// ---------------------------------------------------------------------------
// Kernel 1: fused QKV projection.  [4096 x 512] @ [512 x 512] x3, f16 WMMA,
// f32 accumulate, all operands pre-packed f16.  Q,K stored [b,h,l,d]; V stored
// transposed [b,h,d,l] so attn@V B-fragments are contiguous.
// ---------------------------------------------------------------------------
__global__ __launch_bounds__(256) void qkv_kernel(
    const _Float16* __restrict__ Xh,
    const _Float16* __restrict__ Wqt, const float* __restrict__ bq,
    const _Float16* __restrict__ Wkt, const float* __restrict__ bk,
    const _Float16* __restrict__ Wvt, const float* __restrict__ bv,
    _Float16* __restrict__ Qh, _Float16* __restrict__ Kh,
    _Float16* __restrict__ Vt)
{
    const int lane = threadIdx.x & 31;
    const int wave = threadIdx.x >> 5;
    const int t      = blockIdx.x * 8 + wave;   // 24576 tiles = 256 m * 96 n
    const int tile_m = t / 96;
    const int tile_n = t % 96;
    const int which  = tile_n / 32;             // 0=Q 1=K 2=V
    const int ntile  = tile_n % 32;

    const _Float16* Wt  = (which == 0) ? Wqt : ((which == 1) ? Wkt : Wvt);
    const float*   bvec = (which == 0) ? bq  : ((which == 1) ? bk  : bv);

    const int hlf = lane >> 4;
    const int ml  = lane & 15;
    const int m   = tile_m * 16 + ml;           // A-row this lane supplies
    const int n   = ntile  * 16 + ml;           // B-col this lane supplies

    const _Float16* xrow = Xh + (size_t)m * DD;
    const _Float16* wrow = Wt + (size_t)n * DD;   // transposed: contiguous in k

    v8f c = {};
    for (int kk = 0; kk < DD; kk += 32) {
        // A: elements e -> k = kk + hlf*16 + e  (contiguous run of 16)
        half8 alo = *(const half8*)(xrow + kk + hlf * 16);
        half8 ahi = *(const half8*)(xrow + kk + hlf * 16 + 8);
        // B: elements e -> k = kk + hlf*16 + e  (contiguous run of 16)
        half8 blo = *(const half8*)(wrow + kk + hlf * 16);
        half8 bhi = *(const half8*)(wrow + kk + hlf * 16 + 8);
        half16 a, bfr;
        #pragma unroll
        for (int e = 0; e < 8; ++e) {
            a[e] = alo[e];   a[e + 8] = ahi[e];
            bfr[e] = blo[e]; bfr[e + 8] = bhi[e];
        }
        c = __builtin_amdgcn_wmma_f32_16x16x32_f16(false, a, false, bfr,
                                                   (short)0, c, false, false);
    }

    const float bias_n = bvec[n];
    const int h = n >> 6, d = n & 63;
    #pragma unroll
    for (int r = 0; r < 8; ++r) {
        const int mr = tile_m * 16 + r + 8 * hlf;   // D row
        const int bb = mr >> 10, l = mr & (LSEQ - 1);
        const float val = c[r] + bias_n;
        if (which == 2) {
            Vt[(((size_t)(bb * NH + h)) * DHEAD + d) * LSEQ + l] = (_Float16)val;
        } else {
            _Float16* dst = (which == 0) ? Qh : Kh;
            dst[(((size_t)(bb * NH + h)) * LSEQ + l) * DHEAD + d] = (_Float16)val;
        }
    }
}

// ---------------------------------------------------------------------------
// Kernel 2: fused flash attention with additive bias projection.
// One wave = one (b,h, 16-query) tile; streams 32-key tiles.
// Scores via WMMA f16->f32; bias[b,i,j,0:16] . Wb[:,h] on the VALU (co-issues
// with the matrix pipe; the bias read IS the global bottleneck and is read
// once per head); online softmax via wave32 shfl_xor row reductions; P staged
// through per-wave LDS to re-shape D-layout -> A-layout for attn@V WMMA.
// ---------------------------------------------------------------------------
__global__ __launch_bounds__(256) void attn_kernel(
    const _Float16* __restrict__ Qh, const _Float16* __restrict__ Kh,
    const _Float16* __restrict__ Vt, const float* __restrict__ biasT,
    const float* __restrict__ Wb, _Float16* __restrict__ Yh)
{
    __shared__ __align__(16) _Float16 pbuf[8][16][32];   // per-wave P tile

    const int lane = threadIdx.x & 31;
    const int wave = threadIdx.x >> 5;
    const int t  = blockIdx.x * 8 + wave;   // 2048 = B*H*(L/16)
    const int it = t & 63;
    const int bh = t >> 6;                  // b*H + h
    const int h  = bh & (NH - 1);
    const int bb = bh >> 3;
    const int hlf = lane >> 4, ml = lane & 15;
    const int i0 = it * 16;

    // Wb column for this head (uniform per wave -> scalar loads)
    float wb[16];
    #pragma unroll
    for (int d = 0; d < 16; ++d) wb[d] = Wb[d * NH + h];

    // Q A-fragments (resident for whole j loop): qf[f] covers k = f*32..f*32+31
    half16 qf[2];
    const _Float16* qbase = Qh + ((size_t)bh * LSEQ + i0) * DHEAD;
    #pragma unroll
    for (int f = 0; f < 2; ++f) {
        const _Float16* qrow = qbase + (size_t)ml * DHEAD + f * 32 + hlf * 16;
        half8 lo = *(const half8*)qrow;
        half8 hi = *(const half8*)(qrow + 8);
        #pragma unroll
        for (int e = 0; e < 8; ++e) { qf[f][e] = lo[e]; qf[f][e + 8] = hi[e]; }
    }

    const _Float16* kbase = Kh + (size_t)bh * LSEQ * DHEAD;
    const _Float16* vbase = Vt + (size_t)bh * DHEAD * LSEQ;
    const float*    bbase = biasT + (size_t)bb * LSEQ * LSEQ * 16;

    float mi[8], li[8];
    #pragma unroll
    for (int r = 0; r < 8; ++r) { mi[r] = -1e30f; li[r] = 0.0f; }
    v8f acc[4] = {{}, {}, {}, {}};
    const float scale = 0.125f;   // 1/sqrt(64)

    for (int jt = 0; jt < LSEQ / 32; ++jt) {
        const int j0 = jt * 32;

        if (jt + 1 < LSEQ / 32)   // prefetch next bias stripe (dominant stream)
            __builtin_prefetch(bbase + (((size_t)(i0 + ml)) * LSEQ + j0 + 32) * 16, 0, 1);

        // K B-fragments: kf[jg][f], keys j0+jg*16+0..15, k-chunk f
        half16 kf[2][2];
        #pragma unroll
        for (int jg = 0; jg < 2; ++jg) {
            const _Float16* krow = kbase + (size_t)(j0 + jg * 16 + ml) * DHEAD;
            #pragma unroll
            for (int f = 0; f < 2; ++f) {
                const half8* pv = (const half8*)(krow + f * 32 + hlf * 16);
                half8 lo = pv[0], hi = pv[1];
                #pragma unroll
                for (int e = 0; e < 8; ++e) { kf[jg][f][e] = lo[e]; kf[jg][f][e + 8] = hi[e]; }
            }
        }

        v8f s0 = {}, s1 = {};
        s0 = __builtin_amdgcn_wmma_f32_16x16x32_f16(false, qf[0], false, kf[0][0], (short)0, s0, false, false);
        s0 = __builtin_amdgcn_wmma_f32_16x16x32_f16(false, qf[1], false, kf[0][1], (short)0, s0, false, false);
        s1 = __builtin_amdgcn_wmma_f32_16x16x32_f16(false, qf[0], false, kf[1][0], (short)0, s1, false, false);
        s1 = __builtin_amdgcn_wmma_f32_16x16x32_f16(false, qf[1], false, kf[1][1], (short)0, s1, false, false);

        // Additive bias projection (VALU, co-executes with the WMMA pipe)
        #pragma unroll
        for (int r = 0; r < 8; ++r) {
            const int i = i0 + r + 8 * hlf;
            const float4* bp0 = (const float4*)(bbase + ((size_t)i * LSEQ + j0 + ml) * 16);
            const float4* bp1 = (const float4*)(bbase + ((size_t)i * LSEQ + j0 + 16 + ml) * 16);
            float p0 = 0.0f, p1 = 0.0f;
            #pragma unroll
            for (int q = 0; q < 4; ++q) {
                float4 u = bp0[q], v = bp1[q];
                p0 += u.x * wb[4*q] + u.y * wb[4*q+1] + u.z * wb[4*q+2] + u.w * wb[4*q+3];
                p1 += v.x * wb[4*q] + v.y * wb[4*q+1] + v.z * wb[4*q+2] + v.w * wb[4*q+3];
            }
            s0[r] = s0[r] * scale + p0;
            s1[r] = s1[r] * scale + p1;
        }

        // Online softmax: rows live in 16-lane halves -> xor masks 8,4,2,1
        #pragma unroll
        for (int r = 0; r < 8; ++r) {
            float mx = fmaxf(s0[r], s1[r]);
            #pragma unroll
            for (int off = 8; off >= 1; off >>= 1)
                mx = fmaxf(mx, __shfl_xor(mx, off, 32));
            const float mnew = fmaxf(mi[r], mx);
            const float corr = __expf(mi[r] - mnew);
            const float e0 = __expf(s0[r] - mnew);
            const float e1 = __expf(s1[r] - mnew);
            float rs = e0 + e1;
            #pragma unroll
            for (int off = 8; off >= 1; off >>= 1)
                rs += __shfl_xor(rs, off, 32);
            li[r] = li[r] * corr + rs;
            mi[r] = mnew;
            #pragma unroll
            for (int dc = 0; dc < 4; ++dc) acc[dc][r] *= corr;
            pbuf[wave][r + 8 * hlf][ml]      = (_Float16)e0;   // D layout -> LDS
            pbuf[wave][r + 8 * hlf][ml + 16] = (_Float16)e1;
        }

        // Re-shape P (D layout) -> A fragment via per-wave LDS
        half16 pf;
        {
            const _Float16* prow = &pbuf[wave][ml][0];
            #pragma unroll
            for (int e = 0; e < 8; ++e) {
                pf[e]     = prow[hlf * 16 + e];
                pf[e + 8] = prow[hlf * 16 + 8 + e];
            }
        }

        // attn @ V : contraction over 32 keys, 4 output dim-chunks of 16
        #pragma unroll
        for (int dc = 0; dc < 4; ++dc) {
            const _Float16* vrow = vbase + (size_t)(dc * 16 + ml) * LSEQ + j0 + hlf * 16;
            const half8* pv = (const half8*)vrow;
            half8 lo = pv[0], hi = pv[1];
            half16 vf;
            #pragma unroll
            for (int e = 0; e < 8; ++e) { vf[e] = lo[e]; vf[e + 8] = hi[e]; }
            acc[dc] = __builtin_amdgcn_wmma_f32_16x16x32_f16(false, pf, false, vf,
                                                             (short)0, acc[dc], false, false);
        }
    }

    // Finalize: divide by row sums, store Y[b, i, h*64+d] as f16 for out-proj
    #pragma unroll
    for (int r = 0; r < 8; ++r) {
        const float inv = 1.0f / li[r];
        const int i = i0 + r + 8 * hlf;
        const size_t base = ((size_t)bb * LSEQ + i) * (NH * DHEAD) + h * DHEAD;
        #pragma unroll
        for (int dc = 0; dc < 4; ++dc)
            Yh[base + dc * 16 + ml] = (_Float16)(acc[dc][r] * inv);
    }
}

// ---------------------------------------------------------------------------
// Kernel 3: output projection  [4096 x 512] @ [512 x 512] + bo -> f32 out
// (both operands pre-packed f16; B from transposed Wot, contiguous loads)
// ---------------------------------------------------------------------------
__global__ __launch_bounds__(256) void oproj_kernel(
    const _Float16* __restrict__ Yh, const _Float16* __restrict__ Wot,
    const float* __restrict__ bo, float* __restrict__ out)
{
    const int lane = threadIdx.x & 31;
    const int wave = threadIdx.x >> 5;
    const int t = blockIdx.x * 8 + wave;   // 8192 tiles = 256 x 32
    const int tile_m = t >> 5;
    const int tile_n = t & 31;
    const int hlf = lane >> 4, ml = lane & 15;
    const int m = tile_m * 16 + ml;
    const int n = tile_n * 16 + ml;

    const _Float16* arow = Yh  + (size_t)m * DD;
    const _Float16* wrow = Wot + (size_t)n * DD;

    v8f c = {};
    for (int kk = 0; kk < DD; kk += 32) {
        half8 alo = *(const half8*)(arow + kk + hlf * 16);
        half8 ahi = *(const half8*)(arow + kk + hlf * 16 + 8);
        half8 blo = *(const half8*)(wrow + kk + hlf * 16);
        half8 bhi = *(const half8*)(wrow + kk + hlf * 16 + 8);
        half16 a, bfr;
        #pragma unroll
        for (int e = 0; e < 8; ++e) {
            a[e] = alo[e];   a[e + 8] = ahi[e];
            bfr[e] = blo[e]; bfr[e + 8] = bhi[e];
        }
        c = __builtin_amdgcn_wmma_f32_16x16x32_f16(false, a, false, bfr,
                                                   (short)0, c, false, false);
    }
    const float bn = bo[n];
    #pragma unroll
    for (int r = 0; r < 8; ++r) {
        const int mr = tile_m * 16 + r + 8 * hlf;
        out[(size_t)mr * DD + n] = c[r] + bn;
    }
}

// ---------------------------------------------------------------------------
extern "C" void kernel_launch(void* const* d_in, const int* in_sizes, int n_in,
                              void* d_out, int out_size, void* d_ws, size_t ws_size,
                              hipStream_t stream)
{
    const float* X    = (const float*)d_in[0];
    const float* bias = (const float*)d_in[1];
    const float* Wb   = (const float*)d_in[2];
    const float* Wq   = (const float*)d_in[3];
    const float* bq   = (const float*)d_in[4];
    const float* Wk   = (const float*)d_in[5];
    const float* bk   = (const float*)d_in[6];
    const float* Wv   = (const float*)d_in[7];
    const float* bv   = (const float*)d_in[8];
    const float* Wo   = (const float*)d_in[9];
    const float* bo   = (const float*)d_in[10];
    float* out = (float*)d_out;

    char* ws = (char*)d_ws;                         // 22 MiB used
    _Float16* Qh  = (_Float16*)(ws);                // [B,H,L,64] f16   (4 MiB)
    _Float16* Kh  = (_Float16*)(ws + (4  << 20));   // [B,H,L,64] f16   (4 MiB)
    _Float16* Vt  = (_Float16*)(ws + (8  << 20));   // [B,H,64,L] f16   (4 MiB)
    _Float16* Yh  = (_Float16*)(ws + (12 << 20));   // [B*L, 512] f16   (4 MiB)
    _Float16* Xh  = (_Float16*)(ws + (16 << 20));   // [B*L, 512] f16   (4 MiB)
    _Float16* Wqt = (_Float16*)(ws + (20 << 20));                    // (0.5 MiB)
    _Float16* Wkt = (_Float16*)(ws + (20 << 20) + (512 << 10));      // (0.5 MiB)
    _Float16* Wvt = (_Float16*)(ws + (21 << 20));                    // (0.5 MiB)
    _Float16* Wot = (_Float16*)(ws + (21 << 20) + (512 << 10));      // (0.5 MiB)

    pack_kernel <<<5120, 256, 0, stream>>>(X, Wq, Wk, Wv, Wo,
                                           Xh, Wqt, Wkt, Wvt, Wot);
    qkv_kernel  <<<3072, 256, 0, stream>>>(Xh, Wqt, bq, Wkt, bk, Wvt, bv,
                                           Qh, Kh, Vt);
    attn_kernel <<<256,  256, 0, stream>>>(Qh, Kh, Vt, bias, Wb, Yh);
    oproj_kernel<<<1024, 256, 0, stream>>>(Yh, Wot, bo, out);
}
